// TimeAwareMultiHeadSelfAttention_1322849927616
// MI455X (gfx1250) — matrix operand order repaired and checked by reading, not compiled, not run on hardware
//
#include <hip/hip_runtime.h>

typedef __bf16 bf16_t;
typedef __attribute__((ext_vector_type(16))) __bf16 v16bf;
typedef __attribute__((ext_vector_type(8)))  float  v8f;

#define B_   8
#define L_   1024
#define HID_ 512
#define NH_  8
#define HD_  64
#define NB_  8
#define MT_  8192   // B_*L_

union Frag16 { v16bf v; uint4 q[2]; };

// A-matrix 16x32 bf16 fragment: lane holds row = lane%16; K windows per ISA table.
__device__ __forceinline__ v16bf load_A_frag(const bf16_t* base, int ld) {
  const int lane = threadIdx.x & 31;
  const int row = lane & 15, half = lane >> 4;
  const bf16_t* r = base + row * ld;
  Frag16 u;
  u.q[0] = *(const uint4*)(r + half * 8);
  u.q[1] = *(const uint4*)(r + 16 + half * 8);
  return u.v;
}

// B-matrix 32x16 bf16 fragment from [n][k] row-major storage:
// lane holds col n = lane%16; K windows per ISA table (same as A).
__device__ __forceinline__ v16bf load_B_frag(const bf16_t* base, int ld) {
  const int lane = threadIdx.x & 31;
  const int n = lane & 15, half = lane >> 4;
  const bf16_t* r = base + n * ld;
  Frag16 u;
  u.q[0] = *(const uint4*)(r + half * 8);
  u.q[1] = *(const uint4*)(r + 16 + half * 8);
  return u.v;
}

__device__ __forceinline__ v8f wmma_bf16(v16bf a, v16bf b, v8f c) {
  return __builtin_amdgcn_wmma_f32_16x16x32_bf16(false, a, false, b, (short)0, c,
                                                 false, false);
}

// --- CDNA5 async global->LDS copies (ASYNCcnt) -----------------------------
__device__ __forceinline__ unsigned lds_off_of(const void* p) {
  // flat->LDS mapping uses addr[31:0]; generic ptr low bits == LDS byte offset
  return (unsigned)(uintptr_t)p;
}
__device__ __forceinline__ void async_ld_b128(unsigned lds, int goff,
                                              unsigned long long gbase) {
  asm volatile("global_load_async_to_lds_b128 %0, %1, %2"
               :: "v"(lds), "v"(goff), "s"(gbase) : "memory");
}
__device__ __forceinline__ void async_ld_b64(unsigned lds, int goff,
                                             unsigned long long gbase) {
  asm volatile("global_load_async_to_lds_b64 %0, %1, %2"
               :: "v"(lds), "v"(goff), "s"(gbase) : "memory");
}
__device__ __forceinline__ void wait_async0() {
  asm volatile("s_wait_asynccnt 0x0" ::: "memory");
}
// ---------------------------------------------------------------------------

// --- 16-lane XOR butterfly reductions on the VALU (v_permlane16_b32) -------
#if __has_builtin(__builtin_amdgcn_permlane16)
__device__ __forceinline__ float xr16(float x, unsigned s0, unsigned s1) {
  return __int_as_float(__builtin_amdgcn_permlane16(
      __float_as_int(x), __float_as_int(x), (int)s0, (int)s1, false, false));
}
__device__ __forceinline__ float red16_max(float x) {
  x = fmaxf(x, xr16(x, 0x67452301u, 0xEFCDAB89u));   // xor 1
  x = fmaxf(x, xr16(x, 0x54761032u, 0xDCFE98BAu));   // xor 2
  x = fmaxf(x, xr16(x, 0x32107654u, 0xBA98FEDCu));   // xor 4
  x = fmaxf(x, xr16(x, 0xFEDCBA98u, 0x76543210u));   // xor 8
  return x;
}
__device__ __forceinline__ float red16_sum(float x) {
  x += xr16(x, 0x67452301u, 0xEFCDAB89u);
  x += xr16(x, 0x54761032u, 0xDCFE98BAu);
  x += xr16(x, 0x32107654u, 0xBA98FEDCu);
  x += xr16(x, 0xFEDCBA98u, 0x76543210u);
  return x;
}
#else
__device__ __forceinline__ float red16_max(float x) {
  for (int off = 8; off >= 1; off >>= 1) x = fmaxf(x, __shfl_xor(x, off, 32));
  return x;
}
__device__ __forceinline__ float red16_sum(float x) {
  for (int off = 8; off >= 1; off >>= 1) x += __shfl_xor(x, off, 32);
  return x;
}
#endif
// ---------------------------------------------------------------------------

__global__ __launch_bounds__(256) void cvt_bf16(const float* __restrict__ s,
                                                bf16_t* __restrict__ d, int n4) {
  int i = blockIdx.x * blockDim.x + threadIdx.x;
  int stride = gridDim.x * blockDim.x;
  for (; i < n4; i += stride) {
    float4 f = ((const float4*)s)[i];
    union { bf16_t e[4]; uint2 u; } o;
    o.e[0] = (bf16_t)f.x; o.e[1] = (bf16_t)f.y;
    o.e[2] = (bf16_t)f.z; o.e[3] = (bf16_t)f.w;
    ((uint2*)d)[i] = o.u;
  }
}

// C[M,N] = A[M,K] @ W[N,K]^T + bias, M=8192, N=K=512.
// MODE 0: bf16 output in head-major [B,NH,L,HD]. MODE 1: f32 row-major [M,N].
// Double-buffered LDS staging via async global->LDS copies.
template <int MODE>
__global__ __launch_bounds__(256) void gemm_wmma(const bf16_t* __restrict__ A,
                                                 const bf16_t* __restrict__ W,
                                                 const float* __restrict__ bias,
                                                 void* __restrict__ out) {
  __shared__ bf16_t As[2][128][32];
  __shared__ bf16_t Bs[2][64][32];
  const int t = threadIdx.x;
  const int m0 = blockIdx.x * 128;
  const int n0 = blockIdx.y * 64;
  const int wave = t >> 5, lane = t & 31;
  const int wm = wave >> 1, wn = wave & 1;

  const unsigned long long abase = (unsigned long long)(uintptr_t)A;
  const unsigned long long wbase = (unsigned long long)(uintptr_t)W;
  const unsigned as0 = lds_off_of(&As[0][0][0]);
  const unsigned bs0 = lds_off_of(&Bs[0][0][0]);

  auto stage = [&](int buf, int k0) {
#pragma unroll
    for (int i = 0; i < 2; ++i) {            // A tile: 512 x 16B
      int c = t + i * 256;
      int row = c >> 2, kp = c & 3;
      async_ld_b128(as0 + buf * (128 * 32 * 2) + c * 16,
                    ((m0 + row) * 512 + k0 + kp * 8) * 2, abase);
    }
    {                                        // B tile: 256 x 16B
      int row = t >> 2, kp = t & 3;
      async_ld_b128(bs0 + buf * (64 * 32 * 2) + t * 16,
                    ((n0 + row) * 512 + k0 + kp * 8) * 2, wbase);
    }
  };

  v8f acc[2][2] = {};
  stage(0, 0);
  wait_async0();
  __syncthreads();
  for (int k0 = 0; k0 < 512; k0 += 32) {
    const int cur = (k0 >> 5) & 1;
    if (k0 + 32 < 512) stage(1 - cur, k0 + 32);   // prefetch next tile (async)
#pragma unroll
    for (int ms = 0; ms < 2; ++ms) {
      v16bf a = load_A_frag(&As[cur][wm * 32 + ms * 16][0], 32);
#pragma unroll
      for (int ns = 0; ns < 2; ++ns) {
        v16bf b = load_B_frag(&Bs[cur][wn * 32 + ns * 16][0], 32);
        acc[ms][ns] = wmma_bf16(a, b, acc[ms][ns]);
      }
    }
    wait_async0();
    __syncthreads();
  }

  const int half = lane >> 4, nlo = lane & 15;
#pragma unroll
  for (int ms = 0; ms < 2; ++ms)
#pragma unroll
    for (int ns = 0; ns < 2; ++ns)
#pragma unroll
      for (int vv = 0; vv < 8; ++vv) {
        int mg = m0 + wm * 32 + ms * 16 + vv + half * 8;
        int ng = n0 + wn * 32 + ns * 16 + nlo;
        float val = acc[ms][ns][vv] + bias[ng];
        if (MODE == 0) {
          int bb = mg >> 10, l = mg & 1023, h = ng >> 6, d = ng & 63;
          ((bf16_t*)out)[(((bb * 8 + h) << 10) + l) * 64 + d] = (bf16_t)val;
        } else {
          ((float*)out)[mg * 512 + ng] = val;
        }
      }
}

// Flash attention: block = 16 query rows of one batch, 8 waves = 8 heads.
__global__ __launch_bounds__(256) void attn_kernel(
    const bf16_t* __restrict__ qm, const bf16_t* __restrict__ km,
    const bf16_t* __restrict__ vm, const int* __restrict__ tbm,
    const unsigned char* __restrict__ pad, const float* __restrict__ temb,
    bf16_t* __restrict__ aout) {
  __shared__ int tb_s[16][32];            // shared by all heads (read tbm once)
  __shared__ bf16_t VT[NH_][64][32];      // V chunk transposed, per head
  __shared__ bf16_t Pt[NH_][16][32];      // softmax weights, per head
  __shared__ float te_s[NB_ * NH_];

  const int t = threadIdx.x;
  const int lane = t & 31, h = t >> 5;
  const int b = blockIdx.y, q0 = blockIdx.x * 16;
  const int half = lane >> 4, nlo = lane & 15;

  if (t < NB_ * NH_) te_s[t] = temb[t];

  const unsigned long long tb_base = (unsigned long long)(uintptr_t)tbm;
  const unsigned tbs0 = lds_off_of(&tb_s[0][0]);

  const bf16_t* qbase = qm + ((b * NH_ + h) * L_ + q0) * HD_;
  v16bf qa0 = load_A_frag(qbase, HD_);
  v16bf qa1 = load_A_frag(qbase + 32, HD_);

  float mrow[8], lrow[8];
#pragma unroll
  for (int vv = 0; vv < 8; ++vv) { mrow[vv] = -3.0e38f; lrow[vv] = 0.f; }
  v8f o[4] = {};

  const int nk = (q0 + 16 + 31) >> 5;     // causal: keys <= q0+15
  for (int kb = 0; kb < nk; ++kb) {
    const int kbase = kb * 32;
    __syncthreads();
    {                                      // async stage tb tile 16x32 i32
      int fi = t * 2;
      int r = fi >> 5, cc = fi & 31;
      async_ld_b64(tbs0 + t * 8, ((b * L_ + q0 + r) * L_ + kbase + cc) * 4,
                   tb_base);
    }
    // stage V transposed: exactly 8 jobs/thread, unrolled so all 8
    // global_load_b128 are in flight before the ds_store phase.
#pragma unroll
    for (int i = 0; i < 8; ++i) {
      int c = t + i * 256;
      int hh = c >> 8, rem = c & 255;
      int kc = rem >> 3, db = (rem & 7) * 8;
      union { uint4 q; bf16_t e[8]; } u;
      u.q = *(const uint4*)&vm[((b * NH_ + hh) * L_ + kbase + kc) * HD_ + db];
#pragma unroll
      for (int e = 0; e < 8; ++e) VT[hh][db + e][kc] = u.e[e];
    }
    wait_async0();
    __syncthreads();

    const bf16_t* kbp = km + ((b * NH_ + h) * L_ + kbase) * HD_;
    v8f s0 = {}, s1 = {};
    s0 = wmma_bf16(qa0, load_B_frag(kbp, HD_), s0);
    s0 = wmma_bf16(qa1, load_B_frag(kbp + 32, HD_), s0);
    s1 = wmma_bf16(qa0, load_B_frag(kbp + 16 * HD_, HD_), s1);
    s1 = wmma_bf16(qa1, load_B_frag(kbp + 16 * HD_ + 32, HD_), s1);

    const int col0 = kbase + nlo, col1 = col0 + 16;
    const bool p0 = pad[b * L_ + col0] != 0;
    const bool p1 = pad[b * L_ + col1] != 0;

#pragma unroll
    for (int vv = 0; vv < 8; ++vv) {
      const int r = vv + half * 8;
      const int qg = q0 + r;
      float x0 = (col0 > qg || p0) ? -1e9f
                 : s0[vv] * 0.125f + te_s[tb_s[r][nlo] * NH_ + h];
      float x1 = (col1 > qg || p1) ? -1e9f
                 : s1[vv] * 0.125f + te_s[tb_s[r][nlo + 16] * NH_ + h];
      float mc = red16_max(fmaxf(x0, x1));
      float mnew = fmaxf(mrow[vv], mc);
      float alpha = __expf(mrow[vv] - mnew);
      mrow[vv] = mnew;
      float e0 = __expf(x0 - mnew);
      float e1 = __expf(x1 - mnew);
      float rs = red16_sum(e0 + e1);
      lrow[vv] = lrow[vv] * alpha + rs;
#pragma unroll
      for (int t4 = 0; t4 < 4; ++t4) o[t4][vv] *= alpha;
      Pt[h][r][nlo] = (bf16_t)e0;
      Pt[h][r][nlo + 16] = (bf16_t)e1;
    }
    __syncthreads();

    v16bf pa = load_A_frag(&Pt[h][0][0], 32);
#pragma unroll
    for (int t4 = 0; t4 < 4; ++t4) {
      v16bf vb = load_B_frag(&VT[h][t4 * 16][0], 32);
      o[t4] = wmma_bf16(pa, vb, o[t4]);
    }
  }

#pragma unroll
  for (int vv = 0; vv < 8; ++vv) {
    const int qg = q0 + vv + half * 8;
    const float inv = 1.0f / lrow[vv];
    const bool qp = pad[b * L_ + qg] != 0;
#pragma unroll
    for (int t4 = 0; t4 < 4; ++t4) {
      float val = qp ? 0.f : o[t4][vv] * inv;
      aout[(b * L_ + qg) * HID_ + h * HD_ + t4 * 16 + nlo] = (bf16_t)val;
    }
  }
}

extern "C" void kernel_launch(void* const* d_in, const int* in_sizes, int n_in,
                              void* d_out, int out_size, void* d_ws, size_t ws_size,
                              hipStream_t stream) {
  (void)in_sizes; (void)n_in; (void)out_size; (void)ws_size;
  const float* x = (const float*)d_in[0];
  const int* tbm = (const int*)d_in[1];
  // d_in[2] causal_mask: recomputed analytically
  const unsigned char* pad = (const unsigned char*)d_in[3];
  const float* Wq = (const float*)d_in[4];
  const float* bq = (const float*)d_in[5];
  const float* Wk = (const float*)d_in[6];
  const float* bk = (const float*)d_in[7];
  const float* Wv = (const float*)d_in[8];
  const float* bv = (const float*)d_in[9];
  const float* Wo = (const float*)d_in[10];
  const float* bo = (const float*)d_in[11];
  const float* te = (const float*)d_in[12];
  float* out = (float*)d_out;

  const size_t NX = (size_t)MT_ * HID_;   // 4194304
  const size_t NW = (size_t)HID_ * HID_;  // 262144
  bf16_t* xb  = (bf16_t*)d_ws;
  bf16_t* wqb = xb + NX;
  bf16_t* wkb = wqb + NW;
  bf16_t* wvb = wkb + NW;
  bf16_t* wob = wvb + NW;
  bf16_t* qb  = wob + NW;
  bf16_t* kbf = qb + NX;
  bf16_t* vbf = kbf + NX;
  bf16_t* ab  = vbf + NX;

  cvt_bf16<<<512, 256, 0, stream>>>(x, xb, (int)(NX / 4));
  cvt_bf16<<<128, 256, 0, stream>>>(Wq, wqb, (int)(NW / 4));
  cvt_bf16<<<128, 256, 0, stream>>>(Wk, wkb, (int)(NW / 4));
  cvt_bf16<<<128, 256, 0, stream>>>(Wv, wvb, (int)(NW / 4));
  cvt_bf16<<<128, 256, 0, stream>>>(Wo, wob, (int)(NW / 4));

  dim3 gg(64, 8);
  gemm_wmma<0><<<gg, 256, 0, stream>>>(xb, wqb, bq, (void*)qb);
  gemm_wmma<0><<<gg, 256, 0, stream>>>(xb, wkb, bk, (void*)kbf);
  gemm_wmma<0><<<gg, 256, 0, stream>>>(xb, wvb, bv, (void*)vbf);

  attn_kernel<<<dim3(64, 8), 256, 0, stream>>>(qb, kbf, vbf, tbm, pad, te, ab);

  gemm_wmma<1><<<gg, 256, 0, stream>>>(ab, wob, bo, (void*)out);
}
